// AttentionBlock_13477607375671
// MI455X (gfx1250) — compile-verified, matrix operand order checked by reading, not run on hardware
//
#include <hip/hip_runtime.h>
#include <hip/hip_bf16.h>

typedef _Float16 v8h  __attribute__((ext_vector_type(8)));
typedef _Float16 v16h __attribute__((ext_vector_type(16)));
typedef float    v8f  __attribute__((ext_vector_type(8)));

#define C_DIM 256
#define N_TOK 4096
#define B_DIM 4
#define GROUPS 8

// ---------------------------------------------------------------------------
// WMMA helpers (gfx1250, wave32).  D = A(16x32 f16) * B(32x16 f16) + C(16x16 f32)
// A/Bt fragment: lane row = lane&15, K chunk base kb = (lane>>4)*8,
// holds K = [kb, kb+8) in regs 0..3 and K = [kb+16, kb+24) in regs 4..7.
// D tile: row = (lane>>4)*8 + r, col = lane&15.
// ---------------------------------------------------------------------------
__device__ __forceinline__ v8f wmma16(v16h a, v16h b, v8f c) {
    return __builtin_amdgcn_wmma_f32_16x16x32_f16(
        /*neg_a=*/false, a, /*neg_b=*/false, b,
        /*c_mod=*/(short)0, c, /*reuse_a=*/false, /*reuse_b=*/false);
}

__device__ __forceinline__ v16h frag_ld(const _Float16* p) {
    v8h lo = *reinterpret_cast<const v8h*>(p);
    v8h hi = *reinterpret_cast<const v8h*>(p + 16);
    v16h r;
#pragma unroll
    for (int i = 0; i < 8; ++i) { r[i] = lo[i]; r[i + 8] = hi[i]; }
    return r;
}

__device__ __forceinline__ v8f vzero8() {
    v8f z;
#pragma unroll
    for (int i = 0; i < 8; ++i) z[i] = 0.f;
    return z;
}

// ---------------------------------------------------------------------------
// Kernel 0: f32 -> f16 conversion (weights)
// ---------------------------------------------------------------------------
__global__ void cvt_f16_kernel(const float* __restrict__ in,
                               _Float16* __restrict__ out, int n) {
    int i = blockIdx.x * blockDim.x + threadIdx.x;
    if (i < n) out[i] = (_Float16)in[i];
}

// ---------------------------------------------------------------------------
// Kernel 1: GroupNorm.  x: [B,C,N] f32 -> h: [B,N,C] f16 (token-major for GEMM A)
// One block per (b, group); 32 channels * 4096 pixels reduced per block.
// ---------------------------------------------------------------------------
__global__ void groupnorm_kernel(const float* __restrict__ x,
                                 const float* __restrict__ gamma,
                                 const float* __restrict__ beta,
                                 _Float16* __restrict__ h) {
    const int b = blockIdx.x >> 3;
    const int g = blockIdx.x & 7;
    const int tid = threadIdx.x;
    const int cg = C_DIM / GROUPS;           // 32
    const int base_c = g * cg;
    const int total = cg * N_TOK;            // 131072

    float s = 0.f, s2 = 0.f;
    for (int i = tid; i < total; i += 256) {
        int cc = i >> 12;                    // i / 4096
        int n  = i & 4095;
        float v = x[((size_t)(b * C_DIM + base_c + cc) << 12) + n];
        s += v; s2 += v * v;
    }
    __shared__ float red[256], red2[256];
    red[tid] = s; red2[tid] = s2;
    __syncthreads();
#pragma unroll
    for (int off = 128; off > 0; off >>= 1) {
        if (tid < off) { red[tid] += red[tid + off]; red2[tid] += red2[tid + off]; }
        __syncthreads();
    }
    const float inv_n = 1.0f / (float)total;
    float mu = red[0] * inv_n;
    float var = red2[0] * inv_n - mu * mu;
    float rstd = rsqrtf(var + 1e-5f);

    for (int i = tid; i < total; i += 256) {
        int cc = i >> 12;
        int n  = i & 4095;
        int c  = base_c + cc;
        float v = x[((size_t)(b * C_DIM + c) << 12) + n];
        float y = (v - mu) * rstd * gamma[c] + beta[c];
        h[((size_t)(b * N_TOK + n) << 8) + c] = (_Float16)y;
    }
}

// ---------------------------------------------------------------------------
// Kernel 2: QKV projection.  h:[B,N,C] f16  x  w_qkv:[768,256] f16 (= Bt layout)
// Per-wave 32x32 output tile: 2 A-frags + 2 B-frags -> 4 WMMAs per K-step
// (2x operand reuse vs 16x16 tiling; 1:1 load:WMMA ratio).
// Writes Q,K:[B,N,C], V transposed Vt:[B,C,N].
// ---------------------------------------------------------------------------
__global__ void __launch_bounds__(256) qkv_gemm_kernel(
        const _Float16* __restrict__ h, const _Float16* __restrict__ w,
        const float* __restrict__ bias,
        _Float16* __restrict__ q, _Float16* __restrict__ k,
        _Float16* __restrict__ vt) {
    const int wid  = (blockIdx.x << 3) + (threadIdx.x >> 5);
    const int lane = threadIdx.x & 31;
    const int b   = wid / (128 * 24);
    const int rem = wid % (128 * 24);
    const int mt  = rem / 24;                // 32-row token tile
    const int ot  = rem % 24;                // 32-col output tile (768/32)
    const int rl   = lane & 15;
    const int half = lane >> 4;
    const int kb   = half << 3;
    const int rbase = half << 3;

    const _Float16* pa0 = h + ((size_t)(b * N_TOK + mt * 32 + rl) << 8) + kb;
    const _Float16* pa1 = pa0 + (16 << 8);
    const _Float16* pb0 = w + ((size_t)(ot * 32 + rl) << 8) + kb;
    const _Float16* pb1 = pb0 + (16 << 8);

    v8f acc00 = vzero8(), acc01 = vzero8(), acc10 = vzero8(), acc11 = vzero8();
#pragma unroll
    for (int kc = 0; kc < 8; ++kc) {
        v16h a0 = frag_ld(pa0 + kc * 32);
        v16h a1 = frag_ld(pa1 + kc * 32);
        v16h b0 = frag_ld(pb0 + kc * 32);
        v16h b1 = frag_ld(pb1 + kc * 32);
        acc00 = wmma16(a0, b0, acc00);
        acc01 = wmma16(a0, b1, acc01);
        acc10 = wmma16(a1, b0, acc10);
        acc11 = wmma16(a1, b1, acc11);
    }

#pragma unroll
    for (int oi = 0; oi < 2; ++oi) {
        const int ocol = ot * 32 + oi * 16 + rl;
        const float bv = bias[ocol];
#pragma unroll
        for (int mi = 0; mi < 2; ++mi) {
            const v8f& acc = (mi == 0) ? (oi == 0 ? acc00 : acc01)
                                       : (oi == 0 ? acc10 : acc11);
#pragma unroll
            for (int r = 0; r < 8; ++r) {
                const int n = mt * 32 + mi * 16 + rbase + r;
                const float val = acc[r] + bv;
                if (ot < 8)
                    q[((size_t)(b * N_TOK + n) << 8) + ocol] = (_Float16)val;
                else if (ot < 16)
                    k[((size_t)(b * N_TOK + n) << 8) + (ocol - 256)] = (_Float16)val;
                else
                    vt[((size_t)(b * C_DIM + (ocol - 512)) << 12) + n] = (_Float16)val;
            }
        }
    }
}

// ---------------------------------------------------------------------------
// Kernel 3: flash attention.  Block = (b, 128 query rows), 8 waves x 16 rows.
// Per 32-key step: stage K tile (32x256) + Vt tile (256x32) in LDS, 16 WMMAs
// for S, online softmax, per-wave LDS transpose of P, 16 WMMAs for O += P*V.
// Q fragments pinned in registers (reused across all 128 key blocks).
// ---------------------------------------------------------------------------
__global__ void __launch_bounds__(256, 1) attn_kernel(
        const _Float16* __restrict__ qm, const _Float16* __restrict__ km,
        const _Float16* __restrict__ vm, _Float16* __restrict__ o) {
    __shared__ _Float16 Kt[32 * 256];        // 16 KB: key rows x channels
    __shared__ _Float16 Vt[256 * 32];        // 16 KB: channels x keys
    __shared__ _Float16 Pl[8 * 16 * 32];     //  8 KB: per-wave P staging

    const int tid  = threadIdx.x;
    const int lane = tid & 31;
    const int wv   = tid >> 5;
    const int b  = blockIdx.x >> 5;          // 32 query-blocks per batch
    const int qb = blockIdx.x & 31;
    const int m0 = qb * 128 + wv * 16;
    const int rl = lane & 15, half = lane >> 4;
    const int kb = half << 3, rbase = half << 3;

    // Preload the wave's 16x256 Q tile as 8 A-fragments (64 VGPRs).
    v16h qf[8];
    {
        const _Float16* pq = qm + ((size_t)(b * N_TOK + m0 + rl) << 8) + kb;
#pragma unroll
        for (int kc = 0; kc < 8; ++kc) qf[kc] = frag_ld(pq + kc * 32);
    }

    v8f O[16];
#pragma unroll
    for (int ct = 0; ct < 16; ++ct) O[ct] = vzero8();
    float mrow[8], lrow[8];
#pragma unroll
    for (int r = 0; r < 8; ++r) { mrow[r] = -1e30f; lrow[r] = 0.f; }

    _Float16* pw = Pl + wv * 512;            // this wave's 16x32 P tile

    for (int j = 0; j < 128; ++j) {
        const int n0 = j << 5;
        __syncthreads();
        // Cooperative stage: K tile rows (32 x 256)
        {
            const int row = tid >> 3;
            const int c0  = (tid & 7) << 5;
            const _Float16* src = km + ((size_t)(b * N_TOK + n0 + row) << 8) + c0;
            _Float16* dst = Kt + row * 256 + c0;
#pragma unroll
            for (int i = 0; i < 4; ++i)
                *reinterpret_cast<v8h*>(dst + i * 8) =
                    *reinterpret_cast<const v8h*>(src + i * 8);
            if (j + 1 < 128)
                __builtin_prefetch(src + 32, 0, 1);   // 32 halves past this thread's chunk
        }
        // Cooperative stage: Vt tile (256 x 32)
        {
            const _Float16* src = vm + ((size_t)(b * C_DIM + tid) << 12) + n0;
            _Float16* dst = Vt + tid * 32;
#pragma unroll
            for (int i = 0; i < 4; ++i)
                *reinterpret_cast<v8h*>(dst + i * 8) =
                    *reinterpret_cast<const v8h*>(src + i * 8);
        }
        __syncthreads();

        // S = Q * K^T for 32 keys: two 16x16 D tiles
        v8f s0 = vzero8(), s1 = vzero8();
        const _Float16* pk0 = Kt + rl * 256 + kb;
        const _Float16* pk1 = Kt + (16 + rl) * 256 + kb;
#pragma unroll
        for (int kc = 0; kc < 8; ++kc) {
            s0 = wmma16(qf[kc], frag_ld(pk0 + kc * 32), s0);
            s1 = wmma16(qf[kc], frag_ld(pk1 + kc * 32), s1);
        }

        // Online softmax (scale = 1/sqrt(256) = 0.0625)
        float mn[8], p0[8], p1[8], sc[8], ps[8];
#pragma unroll
        for (int r = 0; r < 8; ++r) {
            float a0 = s0[r] * 0.0625f, a1 = s1[r] * 0.0625f;
            s0[r] = a0; s1[r] = a1;
            mn[r] = fmaxf(a0, a1);
        }
#pragma unroll
        for (int off = 1; off < 16; off <<= 1)
#pragma unroll
            for (int r = 0; r < 8; ++r)
                mn[r] = fmaxf(mn[r], __shfl_xor(mn[r], off, 16));
#pragma unroll
        for (int r = 0; r < 8; ++r) {
            float m2 = fmaxf(mrow[r], mn[r]);
            sc[r] = __expf(mrow[r] - m2);
            mrow[r] = m2;
            p0[r] = __expf(s0[r] - m2);
            p1[r] = __expf(s1[r] - m2);
            ps[r] = p0[r] + p1[r];
        }
#pragma unroll
        for (int off = 1; off < 16; off <<= 1)
#pragma unroll
            for (int r = 0; r < 8; ++r)
                ps[r] += __shfl_xor(ps[r], off, 16);
#pragma unroll
        for (int r = 0; r < 8; ++r) lrow[r] = lrow[r] * sc[r] + ps[r];
#pragma unroll
        for (int ct = 0; ct < 16; ++ct)
#pragma unroll
            for (int r = 0; r < 8; ++r) O[ct][r] *= sc[r];

        // Transpose P (D-layout -> A-layout) through per-wave LDS.
        // Same-wave DS ops are in-order, so no barrier is needed here.
#pragma unroll
        for (int r = 0; r < 8; ++r) {
            pw[(rbase + r) * 32 + rl]      = (_Float16)p0[r];
            pw[(rbase + r) * 32 + 16 + rl] = (_Float16)p1[r];
        }
        v16h pf = frag_ld(pw + rl * 32 + kb);

        // O += P * V  (16 channel tiles, K = 32 keys)
#pragma unroll
        for (int ct = 0; ct < 16; ++ct)
            O[ct] = wmma16(pf, frag_ld(Vt + (ct * 16 + rl) * 32 + kb), O[ct]);
    }

#pragma unroll
    for (int r = 0; r < 8; ++r) lrow[r] = 1.f / lrow[r];
#pragma unroll
    for (int ct = 0; ct < 16; ++ct)
#pragma unroll
        for (int r = 0; r < 8; ++r) {
            const int n = m0 + rbase + r;
            o[((size_t)(b * N_TOK + n) << 8) + ct * 16 + rl] =
                (_Float16)(O[ct][r] * lrow[r]);
        }
}

// ---------------------------------------------------------------------------
// Kernel 4: output projection + bias + residual, 32x32 tiles per wave.
// attn:[B,N,C] f16  x  w_proj:[256,256] f16  ->  out:[B,C,N] f32 (+x)
// ---------------------------------------------------------------------------
__global__ void __launch_bounds__(256) proj_gemm_kernel(
        const _Float16* __restrict__ a, const _Float16* __restrict__ w,
        const float* __restrict__ bias, const float* __restrict__ x,
        float* __restrict__ out) {
    const int wid  = (blockIdx.x << 3) + (threadIdx.x >> 5);
    const int lane = threadIdx.x & 31;
    const int b   = wid / (128 * 8);
    const int rem = wid % (128 * 8);
    const int mt  = rem / 8;                 // 32-row token tile
    const int ct  = rem % 8;                 // 32-col channel tile (256/32)
    const int rl = lane & 15, half = lane >> 4;
    const int kb = half << 3, rbase = half << 3;

    const _Float16* pa0 = a + ((size_t)(b * N_TOK + mt * 32 + rl) << 8) + kb;
    const _Float16* pa1 = pa0 + (16 << 8);
    const _Float16* pb0 = w + ((size_t)(ct * 32 + rl) << 8) + kb;
    const _Float16* pb1 = pb0 + (16 << 8);

    v8f acc00 = vzero8(), acc01 = vzero8(), acc10 = vzero8(), acc11 = vzero8();
#pragma unroll
    for (int kc = 0; kc < 8; ++kc) {
        v16h a0 = frag_ld(pa0 + kc * 32);
        v16h a1 = frag_ld(pa1 + kc * 32);
        v16h b0 = frag_ld(pb0 + kc * 32);
        v16h b1 = frag_ld(pb1 + kc * 32);
        acc00 = wmma16(a0, b0, acc00);
        acc01 = wmma16(a0, b1, acc01);
        acc10 = wmma16(a1, b0, acc10);
        acc11 = wmma16(a1, b1, acc11);
    }

#pragma unroll
    for (int oi = 0; oi < 2; ++oi) {
        const int col = ct * 32 + oi * 16 + rl;
        const float bv = bias[col];
#pragma unroll
        for (int mi = 0; mi < 2; ++mi) {
            const v8f& acc = (mi == 0) ? (oi == 0 ? acc00 : acc01)
                                       : (oi == 0 ? acc10 : acc11);
#pragma unroll
            for (int r = 0; r < 8; ++r) {
                const int n = mt * 32 + mi * 16 + rbase + r;
                const size_t idx = ((size_t)(b * C_DIM + col) << 12) + n;
                out[idx] = x[idx] + acc[r] + bv;
            }
        }
    }
}

// ---------------------------------------------------------------------------
// Host launcher
// ---------------------------------------------------------------------------
extern "C" void kernel_launch(void* const* d_in, const int* in_sizes, int n_in,
                              void* d_out, int out_size, void* d_ws, size_t ws_size,
                              hipStream_t stream) {
    const float* x      = (const float*)d_in[0];
    const float* gamma  = (const float*)d_in[1];
    const float* beta   = (const float*)d_in[2];
    const float* w_qkv  = (const float*)d_in[3];
    const float* b_qkv  = (const float*)d_in[4];
    const float* w_proj = (const float*)d_in[5];
    const float* b_proj = (const float*)d_in[6];

    char* ws = (char*)d_ws;
    size_t off = 0;
    auto carve = [&](size_t bytes) {
        void* p = ws + off;
        off += (bytes + 255) & ~(size_t)255;
        return p;
    };
    const size_t act_bytes = (size_t)B_DIM * N_TOK * C_DIM * sizeof(_Float16); // 8 MB
    _Float16* h16    = (_Float16*)carve(act_bytes);
    _Float16* q16    = (_Float16*)carve(act_bytes);
    _Float16* k16    = (_Float16*)carve(act_bytes);
    _Float16* vt16   = (_Float16*)carve(act_bytes);
    _Float16* attn16 = (_Float16*)carve(act_bytes);
    _Float16* wq16   = (_Float16*)carve((size_t)3 * C_DIM * C_DIM * sizeof(_Float16));
    _Float16* wp16   = (_Float16*)carve((size_t)C_DIM * C_DIM * sizeof(_Float16));
    (void)ws_size; (void)n_in; (void)in_sizes; (void)out_size;

    cvt_f16_kernel<<<(3 * C_DIM * C_DIM + 255) / 256, 256, 0, stream>>>(w_qkv, wq16, 3 * C_DIM * C_DIM);
    cvt_f16_kernel<<<(C_DIM * C_DIM + 255) / 256, 256, 0, stream>>>(w_proj, wp16, C_DIM * C_DIM);

    groupnorm_kernel<<<B_DIM * GROUPS, 256, 0, stream>>>(x, gamma, beta, h16);

    // B * (N/32) * (768/32) tiles, 8 waves/block
    qkv_gemm_kernel<<<(B_DIM * 128 * 24) / 8, 256, 0, stream>>>(h16, wq16, b_qkv, q16, k16, vt16);

    // B * (N/128) blocks
    attn_kernel<<<B_DIM * 32, 256, 0, stream>>>(q16, k16, vt16, attn16);

    // B * (N/32) * (256/32) tiles, 8 waves/block
    proj_gemm_kernel<<<(B_DIM * 128 * 8) / 8, 256, 0, stream>>>(attn16, wp16, b_proj, x, (float*)d_out);
}
